// _NonLocalBlockND_POS_46385646797194
// MI455X (gfx1250) — matrix operand order repaired and checked by reading, not compiled, not run on hardware
//
#include <hip/hip_runtime.h>
#include <hip/hip_bf16.h>
#include <math.h>

// ---------------------------------------------------------------------------
// NonLocal block (pos-encoded self-attention) for MI455X / gfx1250, wave32.
// Flash-attention style fusion; all GEMMs via v_wmma_f32_16x16x32_f16.
// Round 3: O-loop B operands preloaded into distinct registers so the four
// WMMAs issue back-to-back (removes the 4x v_nop WAR-hazard bubbles).
// ---------------------------------------------------------------------------

typedef __attribute__((ext_vector_type(16))) _Float16 v16h;
typedef __attribute__((ext_vector_type(8)))  _Float16 v8h;
typedef __attribute__((ext_vector_type(8)))  float    v8f;

#define NREAL 9409      // 97*97
#define NPAD  9440      // 590*16 == 295*32
#define NT16  590       // NPAD/16
#define CIN   512
#define KMIX  1024      // 2*CIN (x concat pos)
#define C1    256       // theta channels
#define C2    512       // g channels

// --- WMMA helpers (wave32; contiguous k order k = 16*(lane>>4)+elem, same for A and B) -------------

__device__ __forceinline__ v8f wmma16(v16h a, v16h b, v8f c) {
  // D = A(16x32 f16) x B(32x16 f16) + C(16x16 f32)
  return __builtin_amdgcn_wmma_f32_16x16x32_f16(
      /*neg_a=*/false, a, /*neg_b=*/false, b,
      /*c_mod=*/(short)0, c, /*reuse_a=*/false, /*reuse_b=*/false);
}

// A operand: lane L holds row M = L&15.  Elements 0..15 = K (L>>4)*16+0..15.
// `rowk` points at row start + k-step.
__device__ __forceinline__ v16h load_a16(const _Float16* rowk, int khalf) {
  v8h lo = *(const v8h*)(rowk + khalf * 16);       // 16B aligned
  v8h hi = *(const v8h*)(rowk + khalf * 16 + 8);   // 16B aligned
  return __builtin_shufflevector(lo, hi, 0,1,2,3,4,5,6,7,8,9,10,11,12,13,14,15);
}

// --- pack kernels -----------------------------------------------------------

// XT[n][k] = (k<512 ? x[k][n] : pos[k-512][n]), n-major, zero padded rows.
__global__ void pack_xt_kernel(const float* __restrict__ x,
                               const float* __restrict__ pos,
                               _Float16* __restrict__ XT) {
  int idx = blockIdx.x * blockDim.x + threadIdx.x;
  if (idx >= NPAD * KMIX) return;
  int n = idx / KMIX, k = idx % KMIX;
  float v = 0.f;
  if (n < NREAL)
    v = (k < CIN) ? x[(size_t)k * NREAL + n] : pos[(size_t)(k - CIN) * NREAL + n];
  XT[idx] = (_Float16)v;
}

// Fold BN scale into W_theta; bias2 = b*scale + beta.
__global__ void pack_wtheta_kernel(const float* __restrict__ w,
                                   const float* __restrict__ b,
                                   const float* __restrict__ gamma,
                                   const float* __restrict__ beta,
                                   _Float16* __restrict__ wq,
                                   float* __restrict__ bias2) {
  int idx = blockIdx.x * blockDim.x + threadIdx.x;
  if (idx >= C1 * KMIX) return;
  int m = idx / KMIX;
  const float inv = rsqrtf(1.0f + 1e-5f);
  wq[idx] = (_Float16)(w[idx] * gamma[m] * inv);
  if (idx < C1) bias2[idx] = b[idx] * gamma[idx] * inv + beta[idx];
}

__global__ void pack_f16_kernel(const float* __restrict__ s,
                                _Float16* __restrict__ d, int n) {
  int idx = blockIdx.x * blockDim.x + threadIdx.x;
  if (idx < n) d[idx] = (_Float16)s[idx];
}

// --- generic WMMA GEMM:  C[M,Npad] = W[M,K] @ X(n-major)[Npad,K] + bias -----
// EPI 0: theta  -> relu, store transposed tT[n][ch] (f16)
// EPI 1: g      -> store G[ch][n] (f16, n contiguous)
// EPI 2: out    -> store fp32 d_out[ch][n], guard n < NREAL
template <int EPI>
__global__ void gemm16_kernel(const _Float16* __restrict__ W,
                              const _Float16* __restrict__ X,
                              const float* __restrict__ bias,
                              void* __restrict__ outp,
                              int K, int ldx, int totalTiles) {
  int lane = threadIdx.x & 31, wave = threadIdx.x >> 5;
  int tile = blockIdx.x * 8 + wave;
  if (tile >= totalTiles) return;
  int nt = tile % NT16, mt = tile / NT16;
  int khalf = lane >> 4, lidx = lane & 15;

  v8f acc;
#pragma unroll
  for (int r = 0; r < 8; ++r) acc[r] = bias[mt * 16 + r + 8 * khalf];

  const _Float16* wrow = W + (size_t)(mt * 16 + lidx) * K;
  const _Float16* xrow = X + (size_t)(nt * 16 + lidx) * ldx + khalf * 16;
  for (int kk = 0; kk < K; kk += 32) {
    v16h a = load_a16(wrow + kk, khalf);         // A: W rows (row-major)
    v16h b = *(const v16h*)(xrow + kk);          // B: lane=n, elems=K chunk
    acc = wmma16(a, b, acc);
  }

  int n = nt * 16 + lidx;                        // C/D: N = lane&15
#pragma unroll
  for (int r = 0; r < 8; ++r) {
    int ch = mt * 16 + r + 8 * khalf;            // C/D: M = r + 8*(lane>>4)
    float v = acc[r];
    if constexpr (EPI == 0) {
      ((_Float16*)outp)[(size_t)n * C1 + ch] = (_Float16)fmaxf(v, 0.f);
    } else if constexpr (EPI == 1) {
      ((_Float16*)outp)[(size_t)ch * NPAD + n] = (_Float16)v;
    } else {
      if (n < NREAL) ((float*)outp)[(size_t)ch * NREAL + n] = v;
    }
  }
}

// --- fused attention: S = tT_n . t_m / 16 ; online softmax ; O += P.g^T -----
// block = 256 threads (8 waves).  One 16-row n-tile per block.
// tT n-tile lives in LDS (8KB).  Waves 0 and 1 each compute one 16x16 S
// subtile per 32-wide m-step (A re-read from LDS, no big register tile);
// row stats merged through LDS.  Every wave owns 64 channels of c with
// 4 f32 accumulator tiles, A operand = P from LDS, B streamed from g (L2).
__global__ void attn_kernel(const _Float16* __restrict__ tT,   // [NPAD][C1]
                            const _Float16* __restrict__ G,    // [C2][NPAD]
                            _Float16* __restrict__ yT) {       // [NPAD][C2]
  __shared__ alignas(32) _Float16 sT[16 * C1];   // tT rows of this n-tile, 8KB
  __shared__ alignas(32) _Float16 sP[16 * 32];   // P tile, row-major [n][m]
  __shared__ float sMax[2][16];                  // per-subtile row max
  __shared__ float sSum[2][16];                  // per-subtile row sum
  __shared__ float sA[16];                       // alpha / 1/l per row

  int tid = threadIdx.x;
  int lane = tid & 31, wave = tid >> 5;
  int khalf = lane >> 4, lidx = lane & 15;
  int ntile = blockIdx.x;

  // cooperative stage of the 16x256 tT n-tile into LDS (32B per thread)
  {
    int row = tid >> 4, col = (tid & 15) * 16;
    *(v16h*)(sT + row * C1 + col) =
        *(const v16h*)(tT + (size_t)(ntile * 16 + row) * C1 + col);
  }
  __syncthreads();

  float mrun[8], lrun[8];
#pragma unroll
  for (int r = 0; r < 8; ++r) { mrun[r] = -INFINITY; lrun[r] = 0.f; }

  v8f o[4];
#pragma unroll
  for (int t = 0; t < 4; ++t)
#pragma unroll
    for (int r = 0; r < 8; ++r) o[t][r] = 0.f;

  // per-wave g row pointers (advance 32 elements per m-step)
  const _Float16* gp0 = G + (size_t)(wave * 64 + lidx) * NPAD + khalf * 16;

  for (int mb = 0; mb < NPAD; mb += 32) {
    v8f s;
    float alr[8];
    if (wave < 2) {
      // ---- S subtile: rows = this n-tile, cols m = mb + wave*16 + 0..15
#pragma unroll
      for (int r = 0; r < 8; ++r) s[r] = 0.f;
      const _Float16* bp =
          tT + (size_t)(mb + wave * 16 + lidx) * C1 + khalf * 16;
      const _Float16* arow = sT + lidx * C1;     // A from LDS, M = lane&15
#pragma unroll
      for (int kk = 0; kk < 8; ++kk) {           // K = 256 = 8 x 32
        v16h a = load_a16(arow + kk * 32, khalf);
        v16h b = *(const v16h*)(bp + kk * 32);   // B: lane=m, elems=c chunk
        s = wmma16(a, b, s);
      }
      bool ok = (mb + wave * 16 + lidx) < NREAL; // mask padded m columns
#pragma unroll
      for (int r = 0; r < 8; ++r)
        s[r] = ok ? s[r] * 0.0625f : -1e30f;     // * c1^-0.5 = 1/16
      // per-row max within this subtile (row spans a 16-lane group)
#pragma unroll
      for (int r = 0; r < 8; ++r) {
        float v = s[r];
#pragma unroll
        for (int d = 1; d < 16; d <<= 1) v = fmaxf(v, __shfl_xor(v, d, 16));
        if (lidx == 0) sMax[wave][r + 8 * khalf] = v;
      }
    }
    __syncthreads();  // A: subtile maxes visible

    if (wave < 2) {
#pragma unroll
      for (int r = 0; r < 8; ++r) {
        int row = r + 8 * khalf;
        float nm = fmaxf(mrun[r], fmaxf(sMax[0][row], sMax[1][row]));
        float al = __expf(mrun[r] - nm);
        alr[r] = al;
        mrun[r] = nm;                            // replicated in waves 0,1
        float p = __expf(s[r] - nm);
        sP[row * 32 + wave * 16 + lidx] = (_Float16)p;
        float rs = p;
#pragma unroll
        for (int d = 1; d < 16; d <<= 1) rs += __shfl_xor(rs, d, 16);
        if (lidx == 0) sSum[wave][row] = rs;
        if (wave == 0 && lidx == 0) sA[row] = al;
      }
    }
    __syncthreads();  // B: P, alpha, sums visible

    if (wave == 0) {   // running denominator (owned by wave 0)
#pragma unroll
      for (int r = 0; r < 8; ++r) {
        int row = r + 8 * khalf;
        lrun[r] = lrun[r] * alr[r] + sSum[0][row] + sSum[1][row];
      }
    }

    // all waves: preload the four B operands (one clause, distinct regs),
    // rescale O by alpha, then issue the four WMMAs back-to-back
    // (shared A, distinct B/D -> no D->A/B RAW, no WAR NOP bubbles).
    v16h Bg[4];
#pragma unroll
    for (int t = 0; t < 4; ++t) {
      const _Float16* gp = gp0 + (size_t)t * 16 * NPAD + mb;
      __builtin_prefetch((const void*)(gp + 32), 0, 1);  // global_prefetch_b8
      Bg[t] = *(const v16h*)gp;                          // B: lane=c, elems=m
    }

    float alv[8];
#pragma unroll
    for (int r = 0; r < 8; ++r) alv[r] = sA[r + 8 * khalf];
#pragma unroll
    for (int t = 0; t < 4; ++t)
#pragma unroll
      for (int r = 0; r < 8; ++r) o[t][r] *= alv[r];

    v16h Ap = load_a16(sP + lidx * 32, khalf);   // P as A operand

#pragma unroll
    for (int t = 0; t < 4; ++t) o[t] = wmma16(Ap, Bg[t], o[t]);

    __syncthreads();  // C: protect sP/sMax/sA for next iteration
  }

  // epilogue: divide by row sum, store yT[n][c] (f16, n-major)
  if (wave == 0) {
#pragma unroll
    for (int r = 0; r < 8; ++r)
      if (lidx == 0) sA[r + 8 * khalf] = 1.f / lrun[r];
  }
  __syncthreads();
  float liv[8];
#pragma unroll
  for (int r = 0; r < 8; ++r) liv[r] = sA[r + 8 * khalf];
#pragma unroll
  for (int t = 0; t < 4; ++t) {
    int c = wave * 64 + t * 16 + lidx;
#pragma unroll
    for (int r = 0; r < 8; ++r) {
      int n = ntile * 16 + r + 8 * khalf;
      yT[(size_t)n * C2 + c] = (_Float16)(o[t][r] * liv[r]);
    }
  }
}

// --- host-side orchestration ------------------------------------------------

extern "C" void kernel_launch(void* const* d_in, const int* in_sizes, int n_in,
                              void* d_out, int out_size, void* d_ws, size_t ws_size,
                              hipStream_t stream) {
  const float* x      = (const float*)d_in[0];
  const float* pos    = (const float*)d_in[1];
  const float* w_th   = (const float*)d_in[2];
  const float* b_th   = (const float*)d_in[3];
  const float* gamma  = (const float*)d_in[4];
  const float* beta   = (const float*)d_in[5];
  const float* w_g    = (const float*)d_in[6];
  const float* b_g    = (const float*)d_in[7];
  const float* w_out  = (const float*)d_in[8];
  const float* b_out  = (const float*)d_in[9];

  // workspace carve (256B aligned)
  char* ws = (char*)d_ws;
  size_t off = 0;
  auto carve = [&](size_t bytes) {
    char* p = ws + off;
    off = (off + bytes + 255) & ~(size_t)255;
    return p;
  };
  _Float16* XT    = (_Float16*)carve((size_t)NPAD * KMIX * 2); // 19.3 MB
  _Float16* wThq  = (_Float16*)carve((size_t)C1 * KMIX * 2);
  float*    bias2 = (float*)   carve((size_t)C1 * 4);
  _Float16* wGq   = (_Float16*)carve((size_t)C2 * KMIX * 2);
  _Float16* wOq   = (_Float16*)carve((size_t)CIN * C2 * 2);
  _Float16* tT    = (_Float16*)carve((size_t)NPAD * C1 * 2);   // 4.8 MB
  _Float16* Gbuf  = (_Float16*)carve((size_t)C2 * NPAD * 2);   // 9.7 MB
  _Float16* yT    = (_Float16*)carve((size_t)NPAD * C2 * 2);   // 9.7 MB

  // 1) pack activations + weights to f16 (BN folded into W_theta)
  pack_xt_kernel<<<(NPAD * KMIX + 255) / 256, 256, 0, stream>>>(x, pos, XT);
  pack_wtheta_kernel<<<(C1 * KMIX + 255) / 256, 256, 0, stream>>>(
      w_th, b_th, gamma, beta, wThq, bias2);
  pack_f16_kernel<<<(C2 * KMIX + 255) / 256, 256, 0, stream>>>(w_g, wGq, C2 * KMIX);
  pack_f16_kernel<<<(CIN * C2 + 255) / 256, 256, 0, stream>>>(w_out, wOq, CIN * C2);

  // 2) theta: tT[n][256] = relu(BN(W_theta @ x_mix))   (WMMA GEMM, EPI 0)
  {
    int tiles = (C1 / 16) * NT16;
    gemm16_kernel<0><<<(tiles + 7) / 8, 256, 0, stream>>>(
        wThq, XT, bias2, (void*)tT, KMIX, KMIX, tiles);
  }
  // 3) g: G[512][NPAD] = W_g @ x_mix + b_g             (WMMA GEMM, EPI 1)
  {
    int tiles = (C2 / 16) * NT16;
    gemm16_kernel<1><<<(tiles + 7) / 8, 256, 0, stream>>>(
        wGq, XT, b_g, (void*)Gbuf, KMIX, KMIX, tiles);
  }
  // 4) fused attention: yT[n][512] = softmax(tT tTᵀ/16) @ gᵀ
  attn_kernel<<<NT16, 256, 0, stream>>>(tT, Gbuf, yT);

  // 5) out: d_out[512][9409] = W_out @ y + b_out       (WMMA GEMM, EPI 2)
  {
    int tiles = (CIN / 16) * NT16;
    gemm16_kernel<2><<<(tiles + 7) / 8, 256, 0, stream>>>(
        wOq, yT, b_out, d_out, C2, C2, tiles);
  }
}
